// MySchNet_25924422599125
// MI455X (gfx1250) — matrix operand order repaired and checked by reading, not compiled, not run on hardware
//
#include <hip/hip_runtime.h>
#include <hip/hip_bf16.h>
#include <math.h>

// ---------------------------------------------------------------------------
// SchNet-style GNN for MI455X (gfx1250), compile-only tuned.
// - GEMMs on v_wmma_f32_16x16x32_bf16, f32 accumulate.
// - Weights pre-swizzled to WMMA B-fragment order (bf16, zero padded): hot
//   loops are pure b128 streams + WMMA (verified: s_clause'd b128 + wmma).
// - Edge MLP (E x 64 @ 64x128 -> ssp -> E x 128 @ 128x128 -> *Cc) fused in
//   one kernel: the 16x128 intermediate strip never touches HBM; it round-
//   trips through a per-wave 4 KB LDS tile (saves ~1.9 GB total traffic).
// - Transcendentals use hardware TRANS ops (v_exp/v_log/v_cos).
// - Scatter-add via global_atomic_add_f32 into a 10 MB L2-resident buffer.
// ---------------------------------------------------------------------------

typedef __attribute__((ext_vector_type(16))) __bf16 v16bf;
typedef __attribute__((ext_vector_type(4)))  __bf16 v4bf;
typedef __attribute__((ext_vector_type(8)))  float  v8f;

#define HID 128
#define FIL 128
#define NG  50
#define NI  6
#define LOG2F_C 0.69314718055994531f

enum { EPI_NONE = 0, EPI_SSP = 1, EPI_SCALE = 2, EPI_RESID = 3 };

__device__ __forceinline__ float ssp_f(float x) {
  // softplus(x) - log(2) via hardware v_exp_f32 / v_log_f32 (TRANS pipe)
  const float t = __expf(-fabsf(x));
  return fmaxf(x, 0.0f) + __logf(1.0f + t) - LOG2F_C;
}

// A-fragment loads: per ISA 16-bit A 16x32 layout, lane holds row M=lane&15,
// elements 0..7 at K = k0 + e + 8*khalf, elements 8..15 at K = +16.
// pa points at Arow + k0 + 8*khalf -> two contiguous 16B chunks.
__device__ __forceinline__ v16bf load_a(const __bf16* __restrict__ pa) {
  v16bf r;
#pragma unroll
  for (int e = 0; e < 8; ++e) { r[e] = pa[e]; r[e + 8] = pa[e + 16]; }
  return r;
}
__device__ __forceinline__ v16bf load_a(const float* __restrict__ pa) {
  v16bf r;
#pragma unroll
  for (int e = 0; e < 8; ++e) {
    r[e]     = (__bf16)pa[e];
    r[e + 8] = (__bf16)pa[e + 16];
  }
  return r;
}

__device__ __forceinline__ v8f wmma_bf16(const v16bf& a, const v16bf& b, const v8f& c) {
  return __builtin_amdgcn_wmma_f32_16x16x32_bf16(false, a, false, b,
                                                 (short)0, c, false, false);
}

// ---------------------------------------------------------------------------
// Generic WMMA GEMM: C[M x NT*16] = epi(A[M x K] @ B + bias)
// One wave per 16-row strip; NT 16x16 tiles per wave (A fragment reused).
// Bp is pre-swizzled bf16: [K/32][NT][lane(32)][elem(16)].
// ---------------------------------------------------------------------------
template <typename AT, typename CT, int EPI, int NT>
__global__ __launch_bounds__(128)
void wmma_gemm_kernel(const AT* __restrict__ A, const __bf16* __restrict__ Bp,
                      const float* __restrict__ bias,
                      const float* __restrict__ resid,
                      CT* __restrict__ C,
                      int Mtiles, int K)
{
  const int mtile = blockIdx.x * 4 + threadIdx.y;  // wave-uniform
  if (mtile >= Mtiles) return;                     // whole-wave exit
  const int lane  = threadIdx.x;
  const int l15   = lane & 15;
  const int khalf = lane >> 4;
  const int Ncols = NT * 16;

  const AT*     Arow  = A + (size_t)((mtile << 4) + l15) * K;
  const __bf16* Blane = Bp + lane * 16;

  v8f acc[NT];
#pragma unroll
  for (int nt = 0; nt < NT; ++nt) acc[nt] = (v8f){};

  for (int k0 = 0; k0 < K; k0 += 32) {
    const v16bf af = load_a(Arow + k0 + 8 * khalf);
    const __bf16* bbase = Blane + (size_t)(k0 >> 5) * (NT * 512);
#pragma unroll
    for (int nt = 0; nt < NT; ++nt)
      acc[nt] = wmma_bf16(af, *(const v16bf*)(bbase + nt * 512), acc[nt]);
  }

  // C/D layout: VGPR r holds M = r + 8*khalf, N = lane&15
#pragma unroll
  for (int nt = 0; nt < NT; ++nt) {
    const int bcol = nt * 16 + l15;
    const float bv = bias ? bias[bcol] : 0.0f;
#pragma unroll
    for (int r = 0; r < 8; ++r) {
      const int m = (mtile << 4) + r + 8 * khalf;
      float v = acc[nt][r] + bv;
      if (EPI == EPI_SSP)   v = ssp_f(v);
      if (EPI == EPI_RESID) v = v + resid[(size_t)m * Ncols + bcol];
      C[(size_t)m * Ncols + bcol] = (CT)v;
    }
  }
}

// ---------------------------------------------------------------------------
// Fused edge MLP:  W = (ssp(rbf @ W1 + b1) @ W2 + b2) * Cc
// GEMM1 (K=64) -> ssp -> per-wave LDS tile (C-layout -> A-layout transpose)
// -> GEMM2 (K=128) -> scale by Cc -> bf16 out. Intermediate never hits HBM.
// ---------------------------------------------------------------------------
__global__ __launch_bounds__(128)
void edge_mlp_fused_kernel(const __bf16* __restrict__ rbf,
                           const __bf16* __restrict__ Bw1,
                           const float* __restrict__ b1,
                           const __bf16* __restrict__ Bw2,
                           const float* __restrict__ b2,
                           const float* __restrict__ Cc,
                           __bf16* __restrict__ Wout, int Mtiles)
{
  __shared__ __bf16 lds_t1[4][16 * FIL];           // 4 KB per wave
  const int wave  = threadIdx.y;
  const int mtile = blockIdx.x * 4 + wave;         // wave-uniform
  const int lane  = threadIdx.x;
  const int l15   = lane & 15;
  const int khalf = lane >> 4;
  const bool live = (mtile < Mtiles);

  v8f acc[8];
#pragma unroll
  for (int nt = 0; nt < 8; ++nt) acc[nt] = (v8f){};

  if (live) {
    // ---- GEMM1: acc = rbf[16x64] @ W1 ----
    const __bf16* Arow = rbf + (size_t)((mtile << 4) + l15) * 64;
    const __bf16* B1l  = Bw1 + lane * 16;
#pragma unroll
    for (int k0 = 0; k0 < 64; k0 += 32) {
      const v16bf af = load_a(Arow + k0 + 8 * khalf);
      const __bf16* bbase = B1l + (k0 >> 5) * (8 * 512);
#pragma unroll
      for (int nt = 0; nt < 8; ++nt)
        acc[nt] = wmma_bf16(af, *(const v16bf*)(bbase + nt * 512), acc[nt]);
    }
    // ---- epilogue1: ssp -> LDS tile, row-major [m][k] ----
    __bf16* T = lds_t1[wave];
#pragma unroll
    for (int nt = 0; nt < 8; ++nt) {
      const float bv = b1[nt * 16 + l15];
#pragma unroll
      for (int r = 0; r < 8; ++r)
        T[(r + 8 * khalf) * FIL + nt * 16 + l15] = (__bf16)ssp_f(acc[nt][r] + bv);
    }
  }
  __syncthreads();   // ordering fence (terminated waves don't block barriers)
  if (!live) return;

  // ---- GEMM2: acc = T1[16x128] @ W2 (A fragments via contiguous ds loads) --
  const __bf16* Trow = lds_t1[wave] + l15 * FIL;
  const __bf16* B2l  = Bw2 + lane * 16;
#pragma unroll
  for (int nt = 0; nt < 8; ++nt) acc[nt] = (v8f){};
#pragma unroll
  for (int k0 = 0; k0 < FIL; k0 += 32) {
    const v16bf af = load_a(Trow + k0 + 8 * khalf);
    const __bf16* bbase = B2l + (k0 >> 5) * (8 * 512);
#pragma unroll
    for (int nt = 0; nt < 8; ++nt)
      acc[nt] = wmma_bf16(af, *(const v16bf*)(bbase + nt * 512), acc[nt]);
  }
  // ---- epilogue2: (+b2) * Cc[row] -> bf16 ----
#pragma unroll
  for (int nt = 0; nt < 8; ++nt) {
    const int bcol = nt * 16 + l15;
    const float bv = b2[bcol];
#pragma unroll
    for (int r = 0; r < 8; ++r) {
      const int m = (mtile << 4) + r + 8 * khalf;
      Wout[(size_t)m * FIL + bcol] = (__bf16)((acc[nt][r] + bv) * Cc[m]);
    }
  }
}

// ---------------------------------------------------------------------------
// Swizzle fp32 weights (kValid x Ncols, padded to Kpad) into B-fragment order
// ---------------------------------------------------------------------------
__global__ __launch_bounds__(256)
void swizzle_w_kernel(const float* __restrict__ B, __bf16* __restrict__ Bp,
                      int Kpad, int Ncols, int kValid)
{
  const int i = blockIdx.x * blockDim.x + threadIdx.x;
  if (i >= Kpad * Ncols) return;
  const int e      = i & 15;
  const int lane   = (i >> 4) & 31;
  const int frag   = i >> 9;
  const int ntiles = Ncols >> 4;
  const int kt     = frag / ntiles;
  const int nt     = frag - kt * ntiles;
  const int k      = kt * 32 + (e & 7) + 8 * (lane >> 4) + 16 * (e >> 3);
  const int n      = nt * 16 + (lane & 15);
  const float v    = (k < kValid) ? B[(size_t)k * Ncols + n] : 0.0f;
  Bp[i] = (__bf16)v;
}

// ---------------------------------------------------------------------------
// Edge features: distance -> 50 Gaussians (padded to 64, bf16) + cosine cutoff
// ---------------------------------------------------------------------------
__global__ __launch_bounds__(256)
void edge_feats_kernel(const float* __restrict__ pos, const int* __restrict__ ei,
                       int E, __bf16* __restrict__ rbf, float* __restrict__ Cc)
{
  const int e = blockIdx.x * blockDim.x + threadIdx.x;
  if (e >= E) return;
  const int r = ei[e], c = ei[E + e];
  const float dx = pos[r * 3 + 0] - pos[c * 3 + 0];
  const float dy = pos[r * 3 + 1] - pos[c * 3 + 1];
  const float dz = pos[r * 3 + 2] - pos[c * 3 + 2];
  const float d  = sqrtf(dx * dx + dy * dy + dz * dz);
  const float delta = 10.0f / 49.0f;           // CUT/(NG-1)
  const float coeff = -0.5f / (delta * delta);
#pragma unroll 8
  for (int g = 0; g < 64; ++g) {
    const float t = d - (float)g * delta;
    const float v = (g < NG) ? __expf(coeff * t * t) : 0.0f;
    rbf[(size_t)e * 64 + g] = (__bf16)v;
  }
  Cc[e] = 0.5f * (__cosf(d * 0.31415926535897931f) + 1.0f); // pi/CUT
}

__global__ __launch_bounds__(256)
void hinit_kernel(const int* __restrict__ z, const float* __restrict__ emb,
                  float* __restrict__ h, int N)
{
  const int i = blockIdx.x * blockDim.x + threadIdx.x;
  if (i >= N * HID) return;
  h[i] = emb[(size_t)z[i >> 7] * HID + (i & 127)];
}

__global__ __launch_bounds__(256)
void zero_kernel(float* __restrict__ p, int n)
{
  const int i = blockIdx.x * blockDim.x + threadIdx.x;
  if (i < n) p[i] = 0.0f;
}

// m = hW[row] * W ; agg[col] += m   (4 channels per thread)
__global__ __launch_bounds__(256)
void gather_scatter_kernel(const float* __restrict__ hW,
                           const __bf16* __restrict__ W,
                           const int* __restrict__ ei,
                           float* __restrict__ agg, int E)
{
  const int i = blockIdx.x * blockDim.x + threadIdx.x;
  if (i >= E * (HID / 4)) return;
  const int e = i >> 5;
  const int c = (i & 31) << 2;
  const int r  = ei[e];
  const int cl = ei[E + e];
  const float4 hv = *(const float4*)&hW[(size_t)r * HID + c];
  const v4bf   wv = *(const v4bf*)&W[(size_t)e * HID + c];
  float* ap = &agg[(size_t)cl * HID + c];
  unsafeAtomicAdd(ap + 0, hv.x * (float)wv[0]);
  unsafeAtomicAdd(ap + 1, hv.y * (float)wv[1]);
  unsafeAtomicAdd(ap + 2, hv.z * (float)wv[2]);
  unsafeAtomicAdd(ap + 3, hv.w * (float)wv[3]);
}

// out[0] += sum_i ( dot(tout[i,:64], out_w2) + out_b2 )
__global__ __launch_bounds__(256)
void out_reduce_kernel(const float* __restrict__ tout,
                       const float* __restrict__ w2,
                       const float* __restrict__ b2,
                       float* __restrict__ out, int N)
{
  const int i = blockIdx.x * blockDim.x + threadIdx.x;
  if (i >= N) return;
  float a = b2[0];
#pragma unroll 8
  for (int c = 0; c < 64; ++c) a += tout[(size_t)i * 64 + c] * w2[c];
  unsafeAtomicAdd(out, a);
}

// ---------------------------------------------------------------------------
extern "C" void kernel_launch(void* const* d_in, const int* in_sizes, int n_in,
                              void* d_out, int out_size, void* d_ws, size_t ws_size,
                              hipStream_t stream)
{
  const int*   z       = (const int*)  d_in[0];
  const float* pos     = (const float*)d_in[1];
  const int*   ei      = (const int*)  d_in[2];
  const float* emb     = (const float*)d_in[3];
  const float* mlp_w1  = (const float*)d_in[4];
  const float* mlp_b1  = (const float*)d_in[5];
  const float* mlp_w2  = (const float*)d_in[6];
  const float* mlp_b2  = (const float*)d_in[7];
  const float* conv_w1 = (const float*)d_in[8];
  const float* conv_w2 = (const float*)d_in[9];
  const float* conv_b2 = (const float*)d_in[10];
  const float* lin_w   = (const float*)d_in[11];
  const float* lin_b   = (const float*)d_in[12];
  const float* out_w1  = (const float*)d_in[13];
  const float* out_b1  = (const float*)d_in[14];
  const float* out_w2  = (const float*)d_in[15];
  const float* out_b2  = (const float*)d_in[16];

  const int N = in_sizes[0];
  const int E = in_sizes[2] / 2;

  // Workspace carve
  char* wp = (char*)d_ws;
  auto carve = [&](size_t bytes) {
    void* p = (void*)wp;
    wp += (bytes + 255) & ~(size_t)255;
    return p;
  };
  __bf16* rbf  = (__bf16*)carve((size_t)E * 64  * sizeof(__bf16));
  __bf16* Wb   = (__bf16*)carve((size_t)E * FIL * sizeof(__bf16));
  float*  Cc   = (float*) carve((size_t)E * sizeof(float));
  float*  h    = (float*) carve((size_t)N * HID * sizeof(float));
  float*  hW   = (float*) carve((size_t)N * FIL * sizeof(float));
  float*  agg  = (float*) carve((size_t)N * FIL * sizeof(float));
  float*  t2   = (float*) carve((size_t)N * HID * sizeof(float));
  float*  tout = (float*) carve((size_t)N * 64  * sizeof(float));
  // Pre-swizzled bf16 weights (~1 MB total, L2 resident)
  __bf16* Bw1  = (__bf16*)carve((size_t)NI * 64  * FIL * sizeof(__bf16));
  __bf16* Bw2  = (__bf16*)carve((size_t)NI * FIL * FIL * sizeof(__bf16));
  __bf16* Bcw1 = (__bf16*)carve((size_t)NI * HID * FIL * sizeof(__bf16));
  __bf16* Bcw2 = (__bf16*)carve((size_t)NI * FIL * HID * sizeof(__bf16));
  __bf16* Blw  = (__bf16*)carve((size_t)NI * HID * HID * sizeof(__bf16));
  __bf16* Bow1 = (__bf16*)carve((size_t)HID * 64 * sizeof(__bf16));

  const dim3 gblk(32, 4);
  auto gemm_grid = [](int M) { return dim3(((M >> 4) + 3) / 4); };

  // ---- one-time weight swizzles (tiny) ----
  for (int l = 0; l < NI; ++l) {
    swizzle_w_kernel<<<(64 * FIL + 255) / 256, 256, 0, stream>>>(
        mlp_w1 + (size_t)l * NG * FIL, Bw1 + (size_t)l * 64 * FIL, 64, FIL, NG);
    swizzle_w_kernel<<<(FIL * FIL + 255) / 256, 256, 0, stream>>>(
        mlp_w2 + (size_t)l * FIL * FIL, Bw2 + (size_t)l * FIL * FIL, FIL, FIL, FIL);
    swizzle_w_kernel<<<(HID * FIL + 255) / 256, 256, 0, stream>>>(
        conv_w1 + (size_t)l * HID * FIL, Bcw1 + (size_t)l * HID * FIL, HID, FIL, HID);
    swizzle_w_kernel<<<(FIL * HID + 255) / 256, 256, 0, stream>>>(
        conv_w2 + (size_t)l * FIL * HID, Bcw2 + (size_t)l * FIL * HID, FIL, HID, FIL);
    swizzle_w_kernel<<<(HID * HID + 255) / 256, 256, 0, stream>>>(
        lin_w + (size_t)l * HID * HID, Blw + (size_t)l * HID * HID, HID, HID, HID);
  }
  swizzle_w_kernel<<<(HID * 64 + 255) / 256, 256, 0, stream>>>(
      out_w1, Bow1, HID, 64, HID);

  edge_feats_kernel<<<(E + 255) / 256, 256, 0, stream>>>(pos, ei, E, rbf, Cc);
  hinit_kernel<<<(N * HID + 255) / 256, 256, 0, stream>>>(z, emb, h, N);

  for (int l = 0; l < NI; ++l) {
    // Wb = (ssp(rbf @ W1 + b1) @ W2 + b2) * Cc   (fused, LDS intermediate)
    edge_mlp_fused_kernel<<<gemm_grid(E), gblk, 0, stream>>>(
        rbf, Bw1 + (size_t)l * 64 * FIL, mlp_b1 + (size_t)l * FIL,
        Bw2 + (size_t)l * FIL * FIL, mlp_b2 + (size_t)l * FIL,
        Cc, Wb, E >> 4);
    // hW = h @ conv_w1[l]
    wmma_gemm_kernel<float, float, EPI_NONE, 8><<<gemm_grid(N), gblk, 0, stream>>>(
        h, Bcw1 + (size_t)l * HID * FIL, nullptr, nullptr, hW, N >> 4, HID);
    // agg = segment_sum(hW[row] * Wb, col)
    zero_kernel<<<(N * FIL + 255) / 256, 256, 0, stream>>>(agg, N * FIL);
    gather_scatter_kernel<<<(E * (HID / 4) + 255) / 256, 256, 0, stream>>>(
        hW, Wb, ei, agg, E);
    // t2 = ssp(agg @ conv_w2[l] + conv_b2[l])
    wmma_gemm_kernel<float, float, EPI_SSP, 8><<<gemm_grid(N), gblk, 0, stream>>>(
        agg, Bcw2 + (size_t)l * FIL * HID, conv_b2 + (size_t)l * HID,
        nullptr, t2, N >> 4, FIL);
    // h = h + t2 @ lin_w[l] + lin_b[l]
    wmma_gemm_kernel<float, float, EPI_RESID, 8><<<gemm_grid(N), gblk, 0, stream>>>(
        t2, Blw + (size_t)l * HID * HID, lin_b + (size_t)l * HID,
        h, h, N >> 4, HID);
  }

  // tout = ssp(h @ out_w1 + out_b1)   (N x 128 @ 128x64)
  wmma_gemm_kernel<float, float, EPI_SSP, 4><<<gemm_grid(N), gblk, 0, stream>>>(
      h, Bow1, out_b1, nullptr, tout, N >> 4, HID);

  zero_kernel<<<1, 32, 0, stream>>>((float*)d_out, 1);
  out_reduce_kernel<<<(N + 255) / 256, 256, 0, stream>>>(tout, out_w2, out_b2,
                                                         (float*)d_out, N);
}
